// LinearAttention_38611755991498
// MI455X (gfx1250) — matrix operand order repaired and checked by reading, not compiled, not run on hardware
//
#include <hip/hip_runtime.h>

// ---------------- problem constants ----------------
constexpr int B_  = 8;
constexpr int T_  = 2048;
constexpr int H_  = 1024;
constexpr int NH_ = 16;
constexpr int HD_ = 64;
constexpr int M_  = B_ * T_;          // 16384 GEMM rows
constexpr int NSEG = 16;              // scan segments
constexpr int SEGL = T_ / NSEG;       // 128 steps per segment
constexpr int CH   = 32;              // LDS-staged chunk of timesteps

// ---------------- WMMA types (CDNA5 wave32) ----------------
typedef float  v8f   __attribute__((ext_vector_type(8)));
typedef __bf16 v16bf __attribute__((ext_vector_type(16)));
typedef __bf16 v2bf  __attribute__((ext_vector_type(2)));

union FragBF { v16bf v; unsigned long long q[4]; };

// packed f32x2 -> bf16x2; prefer the hardware packed convert, else rely on
// native fptrunc (RNE) which the backend can fuse into v_cvt_pk_bf16_f32.
__device__ __forceinline__ unsigned pk2bf(float lo, float hi) {
#if __has_builtin(__builtin_amdgcn_cvt_pk_bf16_f32)
  union { v2bf v; unsigned u; } cv;
  cv.v = __builtin_amdgcn_cvt_pk_bf16_f32(lo, hi);
  return cv.u;
#else
  union { v2bf v; unsigned u; } cv;
  cv.v[0] = static_cast<__bf16>(lo);
  cv.v[1] = static_cast<__bf16>(hi);
  return cv.u;
#endif
}
__device__ __forceinline__ float sigmoidf_(float x) {
  return 1.0f / (1.0f + __expf(-x));
}

// =====================================================================
// bf16 WMMA GEMM: C(MxN) = A(MxK) @ B(KxN), f32 in memory, f32 accum.
// MODE 0: plain store to Cout.
// MODE 1: qkv epilogue: elu+1 on q/k regions, relayout to (B,NH,T,HD),
//         per-(b,h,t) row sums for the norm.
// Block: 256 thr = 8 waves (4M x 2N), block tile 128x128, BK = 32.
// Staging is software-pipelined: next K-tile's global loads are issued
// before the WMMA block so HBM/L2 latency overlaps compute.
// =====================================================================
constexpr int LDW = 36;  // LDS row stride in bf16 elements (72B, 8B-aligned)

template <int MODE>
__global__ __launch_bounds__(256) void gemm_bf16_wmma(
    const float* __restrict__ A, const float* __restrict__ Bm,
    int K, int N,
    float* __restrict__ Cout,
    float* __restrict__ qh, float* __restrict__ kh, float* __restrict__ vh,
    float* __restrict__ qsum, float* __restrict__ ksum)
{
  __shared__ __align__(16) unsigned short As[128 * LDW];
  __shared__ __align__(16) unsigned short Bs[128 * LDW];

  const int tid   = threadIdx.x;
  const int m0    = blockIdx.x * 128;
  const int n0    = blockIdx.y * 128;
  const int wave  = tid >> 5, lane = tid & 31;
  const int waveM = wave & 3, waveN = wave >> 2;
  const int lrow  = lane & 15, lhalf = lane >> 4;

  v8f acc[2][4];
  const v8f vzero = {0.f, 0.f, 0.f, 0.f, 0.f, 0.f, 0.f, 0.f};
#pragma unroll
  for (int i = 0; i < 2; ++i)
#pragma unroll
    for (int j = 0; j < 4; ++j) acc[i][j] = vzero;

  // staging assignments
  const int a_col  = (tid & 7) * 4;   // A: 8 thr x 4 cols cover BK=32
  const int a_row  = tid >> 3;        // 0..31, x4 passes -> 128 rows
  const int b_n4   = (tid & 31) * 4;  // B: 32 thr x 4 cols cover BN=128
  const int b_kk0  = (tid >> 5) * 2;  // k-pair base, x2 passes -> 32 k

  // pipelined staging registers
  float4 pa[4];
  float4 pb0[2], pb1[2];

  auto load_tiles = [&](int kb) {
    const float* ag = A + (size_t)(m0 + a_row) * K + kb + a_col;
#pragma unroll
    for (int r = 0; r < 4; ++r)
      pa[r] = *(const float4*)(ag + (size_t)r * 32 * K);
#pragma unroll
    for (int p = 0; p < 2; ++p) {
      const float* bg = Bm + (size_t)(kb + b_kk0 + p * 16) * N + n0 + b_n4;
      pb0[p] = *(const float4*)(bg);
      pb1[p] = *(const float4*)(bg + N);
    }
  };
  auto store_tiles = [&]() {
#pragma unroll
    for (int r = 0; r < 4; ++r) {
      unsigned long long pk =
          (unsigned long long)pk2bf(pa[r].x, pa[r].y) |
          ((unsigned long long)pk2bf(pa[r].z, pa[r].w) << 32);
      *(unsigned long long*)&As[(a_row + r * 32) * LDW + a_col] = pk;
    }
#pragma unroll
    for (int p = 0; p < 2; ++p) {
      const int kk = b_kk0 + p * 16;
      *(unsigned*)&Bs[(b_n4 + 0) * LDW + kk] = pk2bf(pb0[p].x, pb1[p].x);
      *(unsigned*)&Bs[(b_n4 + 1) * LDW + kk] = pk2bf(pb0[p].y, pb1[p].y);
      *(unsigned*)&Bs[(b_n4 + 2) * LDW + kk] = pk2bf(pb0[p].z, pb1[p].z);
      *(unsigned*)&Bs[(b_n4 + 3) * LDW + kk] = pk2bf(pb0[p].w, pb1[p].w);
    }
  };

  load_tiles(0);
  for (int kb = 0; kb < K; kb += 32) {
    __syncthreads();
    store_tiles();
    __syncthreads();
    if (kb + 32 < K) load_tiles(kb + 32);  // overlap with WMMA below

    // ---- fragments (ISA 16-bit A/B VGPR layouts) ----
    FragBF af[2], bfr[4];
#pragma unroll
    for (int i = 0; i < 2; ++i) {
      const int m = waveM * 32 + i * 16 + lrow;
      const unsigned long long* pA =
          (const unsigned long long*)&As[m * LDW + lhalf * 8];
      af[i].q[0] = pA[0];
      af[i].q[1] = pA[1];
      const unsigned long long* pA2 =
          (const unsigned long long*)&As[m * LDW + 16 + lhalf * 8];
      af[i].q[2] = pA2[0];
      af[i].q[3] = pA2[1];
    }
#pragma unroll
    for (int j = 0; j < 4; ++j) {
      const int n = waveN * 64 + j * 16 + lrow;
      const unsigned long long* pB =
          (const unsigned long long*)&Bs[n * LDW + lhalf * 8];
      bfr[j].q[0] = pB[0];
      bfr[j].q[1] = pB[1];
      bfr[j].q[2] = pB[4];
      bfr[j].q[3] = pB[5];
    }
#pragma unroll
    for (int i = 0; i < 2; ++i)
#pragma unroll
      for (int j = 0; j < 4; ++j)
        acc[i][j] = __builtin_amdgcn_wmma_f32_16x16x32_bf16(
            false, af[i].v, false, bfr[j].v, (short)0, acc[i][j], false, false);
  }

  // ---------------- epilogue ----------------
  if (MODE == 0) {
#pragma unroll
    for (int i = 0; i < 2; ++i) {
      const int mg0 = m0 + waveM * 32 + i * 16 + 8 * lhalf;
#pragma unroll
      for (int j = 0; j < 4; ++j) {
        const int colg = n0 + waveN * 64 + j * 16 + lrow;
#pragma unroll
        for (int r = 0; r < 8; ++r)
          Cout[(size_t)(mg0 + r) * N + colg] = acc[i][j][r];
      }
    }
  } else {
    // qkv: N = 3072 = [q | k | v], each 1024; wave's 64 cols = exactly 1 head.
    // colbase is wave-uniform: force it scalar so region/head/dst are SGPRs.
    const int colbase = __builtin_amdgcn_readfirstlane(n0 + waveN * 64);
    const int region  = colbase >> 10;          // 0=q 1=k 2=v
    const int hh      = (colbase & 1023) >> 6;  // head
    const int b       = m0 / T_;                // block stays within one b
    float* dst = (region == 0) ? qh : ((region == 1) ? kh : vh);
    const size_t headbase = ((size_t)(b * NH_ + hh)) * T_;
#pragma unroll
    for (int i = 0; i < 2; ++i) {
      float rs[8];
#pragma unroll
      for (int r = 0; r < 8; ++r) rs[r] = 0.f;
      const int mg0 = m0 + waveM * 32 + i * 16 + 8 * lhalf;
#pragma unroll
      for (int j = 0; j < 4; ++j) {
        const int e = j * 16 + lrow;
#pragma unroll
        for (int r = 0; r < 8; ++r) {
          float val = acc[i][j][r];
          if (region < 2) {
            // branchless elu+1: exp path always evaluated, selected by cndmask
            const float ex = __expf(fminf(val, 0.f));
            val = (val > 0.f) ? (val + 1.f) : ex;
          }
          const int t = (mg0 + r) & (T_ - 1);
          dst[(headbase + t) * HD_ + e] = val;
          rs[r] += val;
        }
      }
      if (region < 2) {
        float* sb = (region == 0) ? qsum : ksum;
#pragma unroll
        for (int r = 0; r < 8; ++r) {
          float s = rs[r];
          s += __shfl_xor(s, 1, 32);
          s += __shfl_xor(s, 2, 32);
          s += __shfl_xor(s, 4, 32);
          s += __shfl_xor(s, 8, 32);  // full 16-lane (=64 col) row sum
          rs[r] = s;
        }
        if (lrow == 0) {
#pragma unroll
          for (int r = 0; r < 8; ++r) {
            const int t = (mg0 + r) & (T_ - 1);
            sb[headbase + t] = rs[r];
          }
        }
      }
    }
  }
}

// =====================================================================
// Scan phase 1: per-segment local scan from zero state.
// grid = B*NH*NSEG blocks, 64 threads (thread = state column e).
// Writes un-normalized local outputs to attn and segment-final state F.
// =====================================================================
__global__ __launch_bounds__(64) void scan_phase1(
    const float* __restrict__ qh, const float* __restrict__ kh,
    const float* __restrict__ vh, const float* __restrict__ decay_param,
    float* __restrict__ attn, float* __restrict__ Fbuf)
{
  const int bh  = blockIdx.x / NSEG;
  const int seg = blockIdx.x % NSEG;
  const int h = bh % NH_, b = bh / NH_;
  const int e = threadIdx.x;
  const float dec = sigmoidf_(decay_param[h]);

  float st[HD_];
#pragma unroll
  for (int d = 0; d < HD_; ++d) st[d] = 0.f;

  __shared__ float qs[CH][HD_];
  __shared__ float ks[CH][HD_];

  const size_t segbase = (size_t)bh * T_ * HD_ + (size_t)seg * SEGL * HD_;
  const float* qp = qh + segbase;
  const float* kp = kh + segbase;
  const float* vp = vh + segbase;
  const size_t obase = ((size_t)b * T_ + (size_t)seg * SEGL) * H_ + h * HD_ + e;

  for (int c0 = 0; c0 < SEGL; c0 += CH) {
    __syncthreads();
    for (int tt = 0; tt < CH; ++tt) {
      qs[tt][e] = qp[(size_t)(c0 + tt) * HD_ + e];
      ks[tt][e] = kp[(size_t)(c0 + tt) * HD_ + e];
    }
    __syncthreads();
    for (int tt = 0; tt < CH; ++tt) {
      const int t = c0 + tt;
      const float vte = vp[(size_t)t * HD_ + e];
      float a0 = 0.f, a1 = 0.f, a2 = 0.f, a3 = 0.f;
#pragma unroll
      for (int d = 0; d < HD_; d += 4) {
        float4 kq = *(const float4*)&ks[tt][d];  // uniform addr -> LDS broadcast
        float4 qq = *(const float4*)&qs[tt][d];
        st[d + 0] = st[d + 0] * dec + kq.x * vte; a0 += qq.x * st[d + 0];
        st[d + 1] = st[d + 1] * dec + kq.y * vte; a1 += qq.y * st[d + 1];
        st[d + 2] = st[d + 2] * dec + kq.z * vte; a2 += qq.z * st[d + 2];
        st[d + 3] = st[d + 3] * dec + kq.w * vte; a3 += qq.w * st[d + 3];
      }
      attn[obase + (size_t)t * H_] = (a0 + a1) + (a2 + a3);
    }
  }
  float* Fp = Fbuf + ((size_t)bh * NSEG + seg) * (HD_ * HD_);
#pragma unroll
  for (int d = 0; d < HD_; ++d) Fp[d * HD_ + e] = st[d];
}

// =====================================================================
// Scan phase 2: cross-segment combine S_in(s+1) = d^L * S_in(s) + F_s.
// grid = B*NH blocks, 64 threads. Also emits final new_state.
// =====================================================================
__global__ __launch_bounds__(64) void scan_phase2(
    const float* __restrict__ state_in, const float* __restrict__ Fbuf,
    const float* __restrict__ decay_param,
    float* __restrict__ Sbuf, float* __restrict__ state_out)
{
  const int bh = blockIdx.x;
  const int h = bh % NH_;
  const int e = threadIdx.x;
  const float dec = sigmoidf_(decay_param[h]);
  const float dL  = __powf(dec, (float)SEGL);

  float S[HD_];
  const float* s0 = state_in + (size_t)bh * (HD_ * HD_);
#pragma unroll
  for (int d = 0; d < HD_; ++d) S[d] = s0[d * HD_ + e];

  for (int s = 0; s < NSEG; ++s) {
    float*       Sp = Sbuf + ((size_t)bh * NSEG + s) * (HD_ * HD_);
    const float* Fp = Fbuf + ((size_t)bh * NSEG + s) * (HD_ * HD_);
#pragma unroll
    for (int d = 0; d < HD_; ++d) {
      Sp[d * HD_ + e] = S[d];
      S[d] = S[d] * dL + Fp[d * HD_ + e];
    }
  }
  float* so = state_out + (size_t)bh * (HD_ * HD_);
#pragma unroll
  for (int d = 0; d < HD_; ++d) so[d * HD_ + e] = S[d];
}

// =====================================================================
// Scan phase 3: correction o_t += d^{i+1} * (q_t @ S_in), then normalize.
// grid = B*NH*NSEG blocks, 64 threads.
// =====================================================================
__global__ __launch_bounds__(64) void scan_phase3(
    const float* __restrict__ qh, const float* __restrict__ Sbuf,
    const float* __restrict__ qsum, const float* __restrict__ ksum,
    const float* __restrict__ decay_param, float* __restrict__ attn)
{
  const int bh  = blockIdx.x / NSEG;
  const int seg = blockIdx.x % NSEG;
  const int h = bh % NH_, b = bh / NH_;
  const int e = threadIdx.x;
  const float dec = sigmoidf_(decay_param[h]);

  float S[HD_];
  const float* Sp = Sbuf + ((size_t)bh * NSEG + seg) * (HD_ * HD_);
#pragma unroll
  for (int d = 0; d < HD_; ++d) S[d] = Sp[d * HD_ + e];

  __shared__ float qs[CH][HD_];
  __shared__ float qsn[CH];
  __shared__ float ksn[CH];

  const size_t segbase = (size_t)bh * T_ * HD_ + (size_t)seg * SEGL * HD_;
  const float* qp = qh + segbase;
  const size_t sumbase = (size_t)bh * T_ + (size_t)seg * SEGL;
  const size_t obase = ((size_t)b * T_ + (size_t)seg * SEGL) * H_ + h * HD_ + e;

  float p = dec;  // d^{i+1}, i = local step
  for (int c0 = 0; c0 < SEGL; c0 += CH) {
    __syncthreads();
    for (int tt = 0; tt < CH; ++tt)
      qs[tt][e] = qp[(size_t)(c0 + tt) * HD_ + e];
    if (e < CH) {
      qsn[e] = qsum[sumbase + c0 + e];
      ksn[e] = ksum[sumbase + c0 + e];
    }
    __syncthreads();
    for (int tt = 0; tt < CH; ++tt) {
      float a0 = 0.f, a1 = 0.f, a2 = 0.f, a3 = 0.f;
#pragma unroll
      for (int d = 0; d < HD_; d += 4) {
        float4 qq = *(const float4*)&qs[tt][d];
        a0 += qq.x * S[d + 0];
        a1 += qq.y * S[d + 1];
        a2 += qq.z * S[d + 2];
        a3 += qq.w * S[d + 3];
      }
      const float corr = (a0 + a1) + (a2 + a3);
      const float nrm  = fmaxf(qsn[tt] * ksn[tt], 1e-6f);
      const size_t oi = obase + (size_t)(c0 + tt) * H_;
      attn[oi] = (attn[oi] + corr * p) / nrm;
      p *= dec;
    }
  }
}

// =====================================================================
// launch
// =====================================================================
extern "C" void kernel_launch(void* const* d_in, const int* in_sizes, int n_in,
                              void* d_out, int out_size, void* d_ws, size_t ws_size,
                              hipStream_t stream) {
  (void)in_sizes; (void)n_in; (void)out_size; (void)ws_size;
  const float* x      = (const float*)d_in[0];
  const float* state0 = (const float*)d_in[1];
  const float* Wqkv   = (const float*)d_in[2];
  const float* Wout   = (const float*)d_in[3];
  const float* decay  = (const float*)d_in[4];

  float* out       = (float*)d_out;
  float* state_out = out + (size_t)B_ * T_ * H_;

  float* ws = (float*)d_ws;
  size_t off = 0;
  float* qh   = ws + off; off += (size_t)B_ * NH_ * T_ * HD_;
  float* kh   = ws + off; off += (size_t)B_ * NH_ * T_ * HD_;
  float* vh   = ws + off; off += (size_t)B_ * NH_ * T_ * HD_;
  float* attn = ws + off; off += (size_t)B_ * T_ * H_;
  float* Fbuf = ws + off; off += (size_t)B_ * NH_ * NSEG * HD_ * HD_;
  float* Sbuf = ws + off; off += (size_t)B_ * NH_ * NSEG * HD_ * HD_;
  float* qsb  = ws + off; off += (size_t)B_ * NH_ * T_;
  float* ksb  = ws + off; off += (size_t)B_ * NH_ * T_;

  // 1) qkv projection + elu+1 + head-major relayout + norm row-sums
  dim3 g1(M_ / 128, (3 * H_) / 128);
  gemm_bf16_wmma<1><<<g1, 256, 0, stream>>>(x, Wqkv, H_, 3 * H_,
      nullptr, qh, kh, vh, qsb, ksb);

  // 2) segment-parallel linear scan (3 phases)
  scan_phase1<<<B_ * NH_ * NSEG, 64, 0, stream>>>(qh, kh, vh, decay, attn, Fbuf);
  scan_phase2<<<B_ * NH_, 64, 0, stream>>>(state0, Fbuf, decay, Sbuf, state_out);
  scan_phase3<<<B_ * NH_ * NSEG, 64, 0, stream>>>(qh, Sbuf, qsb, ksb, decay, attn);

  // 3) output projection
  dim3 g2(M_ / 128, H_ / 128);
  gemm_bf16_wmma<0><<<g2, 256, 0, stream>>>(attn, Wout, H_, H_,
      out, nullptr, nullptr, nullptr, nullptr, nullptr);
}
